// RecurrentBaseline_88845693485500
// MI455X (gfx1250) — matrix operand order repaired and checked by reading, not compiled, not run on hardware
//
#include <hip/hip_runtime.h>
#include <hip/hip_bf16.h>
#include <stdint.h>
#include <stddef.h>

// Problem constants (match reference)
#define B_   128
#define A_   10
#define T_   64
#define DM_  4
#define H_   128
#define L_   2
#define DH_  1280          // A_*H_
#define G4_  5120          // 4*DH_
#define NT_  63            // T_-1

typedef __bf16 bf16_t;
typedef __attribute__((ext_vector_type(16))) __bf16 v16bf;
typedef __attribute__((ext_vector_type(8)))  __bf16 v8bf;
typedef __attribute__((ext_vector_type(8)))  float  v8f;

// ---------------------------------------------------------------------------
// WMMA helpers (CDNA5 v_wmma_f32_16x16x32_bf16, wave32)
// ---------------------------------------------------------------------------
__device__ inline v8f wmma_bf16(v16bf a, v16bf b, v8f c) {
  // (neg_a, A, neg_b, B, c_mod, C, reuse_a, reuse_b)
  return __builtin_amdgcn_wmma_f32_16x16x32_bf16(false, a, false, b, (short)0, c,
                                                 false, false);
}

__device__ inline v8f zero8() {
  v8f z = {0.f, 0.f, 0.f, 0.f, 0.f, 0.f, 0.f, 0.f};
  return z;
}

// A fragment: 16x32 bf16 tile of row-major A at (m0, k0), row stride lda.
// lanes 0-15: rows m0..m0+15, K = k0+[0..7] and k0+[16..23]
// lanes 16-31: same rows,      K = k0+[8..15] and k0+[24..31]
__device__ inline v16bf load_a_frag(const bf16_t* A, int lda, int m0, int k0, int lane) {
  const int row  = m0 + (lane & 15);
  const int koff = k0 + ((lane >> 4) << 3);
  const bf16_t* p = A + (size_t)row * lda + koff;
  union { v16bf v; v8bf h[2]; } u;
  u.h[0] = *(const v8bf*)(p);
  u.h[1] = *(const v8bf*)(p + 16);
  return u.v;
}

// B fragment: 32x16 bf16 tile. W stored as [N][K] row-major (i.e. B = W^T).
// lanes 0-15: col n0+lane, K = k0+[0..7] and k0+[16..23];  lanes 16-31: col n0+lane-16, K = k0+[8..15] and k0+[24..31]
__device__ inline v16bf load_b_frag(const bf16_t* W, int ldk, int n0, int k0, int lane) {
  const int col  = n0 + (lane & 15);
  const int koff = k0 + ((lane >> 4) << 3);
  const bf16_t* p = W + (size_t)col * ldk + koff;
  union { v16bf v; v8bf h[2]; } u;
  u.h[0] = *(const v8bf*)(p);
  u.h[1] = *(const v8bf*)(p + 16);
  return u.v;
}

__device__ inline float sigmf(float x) { return 1.0f / (1.0f + __expf(-x)); }

// ---------------------------------------------------------------------------
// One-time packing kernels (run each launch; cheap vs 63 recurrent steps)
// ---------------------------------------------------------------------------
__global__ void pack_bf16_kernel(const float* __restrict__ in, bf16_t* __restrict__ out,
                                 size_t n) {
  for (size_t i = (size_t)blockIdx.x * blockDim.x + threadIdx.x; i < n;
       i += (size_t)gridDim.x * blockDim.x)
    out[i] = (bf16_t)in[i];
}

// out[n][k] = in[k][n]  (bf16), rows n>=N zero-padded up to padN
__global__ void pack_transpose_kernel(const float* __restrict__ in, bf16_t* __restrict__ out,
                                      int K, int N, int padN) {
  const size_t total = (size_t)padN * K;
  for (size_t i = (size_t)blockIdx.x * blockDim.x + threadIdx.x; i < total;
       i += (size_t)gridDim.x * blockDim.x) {
    int n = (int)(i / K), k = (int)(i - (size_t)n * K);
    out[i] = (bf16_t)((n < N) ? in[(size_t)k * N + n] : 0.0f);
  }
}

__global__ void bias_sum_kernel(const float* __restrict__ a, const float* __restrict__ b,
                                float* __restrict__ o, int n) {
  int i = blockIdx.x * blockDim.x + threadIdx.x;
  if (i < n) o[i] = a[i] + b[i];
}

__global__ void zero_kernel(uint32_t* __restrict__ p, size_t ndw) {
  for (size_t i = (size_t)blockIdx.x * blockDim.x + threadIdx.x; i < ndw;
       i += (size_t)gridDim.x * blockDim.x)
    p[i] = 0u;
}

// ---------------------------------------------------------------------------
// Encoder with fused burn-in select:
//   ins = (t <= burn) ? inputs[:,:,t,:] : cur_in        (cur_in updated in gt case)
//   e1  = relu(ins @ enc_w1 + b1)   (VALU, K=4 -> LDS bf16)
//   x   = relu(e1 @ enc_w2 + b2)    (WMMA 128x128x128 per WG)
// grid = 10 WGs of 256 threads; WG g handles (b,a) rows [g*128, g*128+128)
// Waves arranged 2(M:64 rows) x 4(N:32 cols): alpha=4, beta=2 per wave.
// ---------------------------------------------------------------------------
__global__ __launch_bounds__(256) void encoder_kernel(
    const float* __restrict__ gt,       // [128][10][64][4] ground-truth inputs
    const int*  __restrict__ burn,      // scalar burn_in_steps
    float* __restrict__ cur_in,         // [128][10][4] in/out
    const float* __restrict__ ew1,      // [4][128]
    const float* __restrict__ eb1,      // [128]
    const bf16_t* __restrict__ ew2t,    // [128][128]  (= enc_w2^T, [N][K])
    const float* __restrict__ eb2,      // [128]
    bf16_t* __restrict__ xout,          // [128][1280]
    int t)
{
  __shared__ bf16_t e1[128 * 136];      // padded row stride 136 (keeps 16B alignment)
  const int tid = threadIdx.x;
  const int r0  = blockIdx.x * 128;

  { // phase 1: input select + tiny K=4 GEMM + relu -> LDS (bf16)
    const int row = tid >> 1;
    const int cb  = (tid & 1) * 64;
    const int gr  = r0 + row;
    float v0, v1, v2, v3;
    if (t <= *burn) {
      const int b = gr / A_, a = gr - b * A_;
      const float* src = gt + (size_t)b * (A_ * T_ * DM_) + (size_t)a * (T_ * DM_) + t * DM_;
      v0 = src[0]; v1 = src[1]; v2 = src[2]; v3 = src[3];
      if ((tid & 1) == 0) {  // each row written by exactly one thread
        cur_in[gr * 4 + 0] = v0; cur_in[gr * 4 + 1] = v1;
        cur_in[gr * 4 + 2] = v2; cur_in[gr * 4 + 3] = v3;
      }
    } else {  // no writes to cur_in happen anywhere in this branch -> no race
      v0 = cur_in[gr * 4 + 0]; v1 = cur_in[gr * 4 + 1];
      v2 = cur_in[gr * 4 + 2]; v3 = cur_in[gr * 4 + 3];
    }
    #pragma unroll 4
    for (int cc = 0; cc < 64; ++cc) {
      const int c = cb + cc;
      float acc = eb1[c] + v0 * ew1[c] + v1 * ew1[128 + c] + v2 * ew1[256 + c] +
                  v3 * ew1[384 + c];
      e1[row * 136 + c] = (bf16_t)fmaxf(acc, 0.0f);
    }
  }
  __syncthreads();

  // phase 2: wave tiling 2x4 -> 64 rows x 32 cols per wave (alpha=4, beta=2)
  const int lane = tid & 31;
  const int w    = tid >> 5;
  const int m0   = (w >> 2) * 64;
  const int n0   = (w & 3) * 32;

  v8f acc[4][2];
  #pragma unroll
  for (int mt = 0; mt < 4; ++mt)
    #pragma unroll
    for (int j = 0; j < 2; ++j) acc[mt][j] = zero8();

  #pragma unroll
  for (int kt = 0; kt < 4; ++kt) {
    const int k0 = kt * 32;
    v16bf a[4];
    #pragma unroll
    for (int mt = 0; mt < 4; ++mt) a[mt] = load_a_frag(e1, 136, m0 + mt * 16, k0, lane);
    #pragma unroll
    for (int j = 0; j < 2; ++j) {
      const v16bf b = load_b_frag(ew2t, 128, n0 + j * 16, k0, lane);
      #pragma unroll
      for (int mt = 0; mt < 4; ++mt) acc[mt][j] = wmma_bf16(a[mt], b, acc[mt][j]);
    }
  }

  const int chalf = (lane >> 4) << 3;
  const int ncol  = lane & 15;
  #pragma unroll
  for (int j = 0; j < 2; ++j) {
    const int h = n0 + j * 16 + ncol;
    const float bh = eb2[h];
    #pragma unroll
    for (int mt = 0; mt < 4; ++mt)
      #pragma unroll
      for (int r = 0; r < 8; ++r) {
        const int gr = r0 + m0 + mt * 16 + chalf + r;
        const int b  = gr / A_;
        const int a  = gr - b * A_;
        xout[(size_t)b * DH_ + a * H_ + h] = (bf16_t)fmaxf(acc[mt][j][r] + bh, 0.0f);
      }
  }
}

// ---------------------------------------------------------------------------
// LSTM layer: gates = x @ Wih^T + h @ Whh^T + (b_ih+b_hh); fused gate pointwise.
// grid = 20 WGs x 256 thr; WG owns gate-columns [c0, c0+64) for ALL 4 gates.
// Waves arranged 2(M) x 4(N): wave owns 64 rows x 16 cols, all 4 gates
//   -> alpha=4 A-tiles + beta=4 B-tiles = 8 fragment loads per 16 WMMAs
//      (vs 17 before: halves WGP$/L2 fragment traffic, XDL-bound loop).
// Gate pointwise stays entirely in registers (identical lane mapping).
// ---------------------------------------------------------------------------
__global__ __launch_bounds__(256) void lstm_layer_kernel(
    const bf16_t* __restrict__ x,      // [128][1280]
    const bf16_t* __restrict__ hprev,  // [128][1280]
    const bf16_t* __restrict__ wih,    // [5120][1280] bf16 ([N][K])
    const bf16_t* __restrict__ whh,    // [5120][1280]
    const float* __restrict__ bsum,    // [5120] = b_ih + b_hh
    float* __restrict__ c,             // [128][1280] fp32, in/out
    bf16_t* __restrict__ hnew)         // [128][1280]
{
  const int lane = threadIdx.x & 31;
  const int w    = threadIdx.x >> 5;
  const int m0   = (w >> 2) * 64;                 // 64-row slab
  const int gc   = blockIdx.x * 64 + (w & 3) * 16; // 16 gate-cols, shared by 4 gates

  v8f acc[4][4];                                  // [row-tile][gate]
  #pragma unroll
  for (int mt = 0; mt < 4; ++mt)
    #pragma unroll
    for (int g = 0; g < 4; ++g) acc[mt][g] = zero8();

  // K-loop 1: input contribution (x, W_ih)
  #pragma unroll 1
  for (int kt = 0; kt < DH_ / 32; ++kt) {
    const int k0 = kt * 32;
    v16bf a[4];
    #pragma unroll
    for (int mt = 0; mt < 4; ++mt) a[mt] = load_a_frag(x, DH_, m0 + mt * 16, k0, lane);
    #pragma unroll
    for (int g = 0; g < 4; ++g) {
      const v16bf b = load_b_frag(wih, DH_, g * DH_ + gc, k0, lane);
      #pragma unroll
      for (int mt = 0; mt < 4; ++mt) acc[mt][g] = wmma_bf16(a[mt], b, acc[mt][g]);
    }
  }
  // K-loop 2: recurrent contribution (h_prev, W_hh)
  #pragma unroll 1
  for (int kt = 0; kt < DH_ / 32; ++kt) {
    const int k0 = kt * 32;
    v16bf a[4];
    #pragma unroll
    for (int mt = 0; mt < 4; ++mt) a[mt] = load_a_frag(hprev, DH_, m0 + mt * 16, k0, lane);
    #pragma unroll
    for (int g = 0; g < 4; ++g) {
      const v16bf b = load_b_frag(whh, DH_, g * DH_ + gc, k0, lane);
      #pragma unroll
      for (int mt = 0; mt < 4; ++mt) acc[mt][g] = wmma_bf16(a[mt], b, acc[mt][g]);
    }
  }

  // fused gate pointwise (torch order i,f,g,o) -- fully in registers
  const int chalf = (lane >> 4) << 3;
  const int col   = gc + (lane & 15);
  const float bi = bsum[col];
  const float bf = bsum[DH_ + col];
  const float bg = bsum[2 * DH_ + col];
  const float bo = bsum[3 * DH_ + col];
  #pragma unroll
  for (int mt = 0; mt < 4; ++mt) {
    #pragma unroll
    for (int r = 0; r < 8; ++r) {
      const int row = m0 + mt * 16 + chalf + r;
      const size_t idx = (size_t)row * DH_ + col;
      const float iv = sigmf(acc[mt][0][r] + bi);
      const float fv = sigmf(acc[mt][1][r] + bf);
      const float gv = tanhf(acc[mt][2][r] + bg);
      const float ov = sigmf(acc[mt][3][r] + bo);
      const float cn = fv * c[idx] + iv * gv;
      c[idx] = cn;
      hnew[idx] = (bf16_t)(ov * tanhf(cn));
    }
  }
}

// ---------------------------------------------------------------------------
// Decoder GEMM 1: y1 = relu(h1 @ dec_w1 + dec_b1)   (M=128,N=1280,K=1280)
// grid = 10 WGs x 128 cols; waves 2(M:64 rows) x 4(N:32 cols); alpha=4, beta=2
// ---------------------------------------------------------------------------
__global__ __launch_bounds__(256) void dec1_kernel(
    const bf16_t* __restrict__ A,     // [128][1280]
    const bf16_t* __restrict__ W,     // [1280][1280] ([N][K])
    const float* __restrict__ bias,   // [1280]
    bf16_t* __restrict__ out)         // [128][1280]
{
  const int lane = threadIdx.x & 31;
  const int w    = threadIdx.x >> 5;
  const int m0   = (w >> 2) * 64;
  const int n0   = blockIdx.x * 128 + (w & 3) * 32;

  v8f acc[4][2];
  #pragma unroll
  for (int mt = 0; mt < 4; ++mt)
    #pragma unroll
    for (int j = 0; j < 2; ++j) acc[mt][j] = zero8();

  #pragma unroll 1
  for (int kt = 0; kt < DH_ / 32; ++kt) {
    const int k0 = kt * 32;
    v16bf a[4];
    #pragma unroll
    for (int mt = 0; mt < 4; ++mt) a[mt] = load_a_frag(A, DH_, m0 + mt * 16, k0, lane);
    #pragma unroll
    for (int j = 0; j < 2; ++j) {
      const v16bf b = load_b_frag(W, DH_, n0 + j * 16, k0, lane);
      #pragma unroll
      for (int mt = 0; mt < 4; ++mt) acc[mt][j] = wmma_bf16(a[mt], b, acc[mt][j]);
    }
  }

  const int chalf = (lane >> 4) << 3;
  const int ncol  = lane & 15;
  #pragma unroll
  for (int j = 0; j < 2; ++j) {
    const int col = n0 + j * 16 + ncol;
    const float bc = bias[col];
    #pragma unroll
    for (int mt = 0; mt < 4; ++mt)
      #pragma unroll
      for (int r = 0; r < 8; ++r)
        out[(size_t)(m0 + mt * 16 + chalf + r) * DH_ + col] =
            (bf16_t)fmaxf(acc[mt][j][r] + bc, 0.0f);
  }
}

// ---------------------------------------------------------------------------
// Decoder GEMM 2 + residual: delta = y1 @ dec_w2 + b2; out = ins + delta.
// N=40 padded to 48; one WG; writes d_out time-slice t AND cur_in (next step in).
// ---------------------------------------------------------------------------
__global__ __launch_bounds__(256) void dec2_kernel(
    const bf16_t* __restrict__ y1,    // [128][1280]
    const bf16_t* __restrict__ W,     // [48][1280] ([N][K], rows 40..47 zero)
    const float* __restrict__ b2,     // [40]
    float* __restrict__ cur_in,       // [128][40] in/out
    float* __restrict__ out,          // [128][10][63][4]
    int t)
{
  const int lane = threadIdx.x & 31;
  const int m0   = (threadIdx.x >> 5) * 16;

  v8f acc[3];
  #pragma unroll
  for (int j = 0; j < 3; ++j) acc[j] = zero8();

  #pragma unroll 1
  for (int kt = 0; kt < DH_ / 32; ++kt) {
    const int k0 = kt * 32;
    const v16bf a = load_a_frag(y1, DH_, m0, k0, lane);
    #pragma unroll
    for (int j = 0; j < 3; ++j) {
      const v16bf b = load_b_frag(W, DH_, j * 16, k0, lane);
      acc[j] = wmma_bf16(a, b, acc[j]);
    }
  }

  const int rbase = m0 + ((lane >> 4) << 3);
  const int ncol  = lane & 15;
  #pragma unroll
  for (int j = 0; j < 3; ++j) {
    const int col = j * 16 + ncol;
    if (col < A_ * DM_) {
      const float bc = b2[col];
      const int a  = col >> 2;
      const int dm = col & 3;
      #pragma unroll
      for (int r = 0; r < 8; ++r) {
        const int row = rbase + r;
        const float ins = cur_in[row * (A_ * DM_) + col];
        const float o = ins + acc[j][r] + bc;
        cur_in[row * (A_ * DM_) + col] = o;                       // next-step input
        out[(size_t)row * (A_ * NT_ * DM_) + (size_t)a * (NT_ * DM_) + t * DM_ + dm] = o;
      }
    }
  }
}

// ---------------------------------------------------------------------------
// Host-side orchestration
// ---------------------------------------------------------------------------
extern "C" void kernel_launch(void* const* d_in, const int* in_sizes, int n_in,
                              void* d_out, int out_size, void* d_ws, size_t ws_size,
                              hipStream_t stream) {
  const float* inputs = (const float*)d_in[0];
  const float* enc_w1 = (const float*)d_in[1];
  const float* enc_b1 = (const float*)d_in[2];
  const float* enc_w2 = (const float*)d_in[3];
  const float* enc_b2 = (const float*)d_in[4];
  const float* w_ih   = (const float*)d_in[5];
  const float* w_hh   = (const float*)d_in[6];
  const float* b_ih   = (const float*)d_in[7];
  const float* b_hh   = (const float*)d_in[8];
  const float* dec_w1 = (const float*)d_in[9];
  const float* dec_b1 = (const float*)d_in[10];
  const float* dec_w2 = (const float*)d_in[11];
  const float* dec_b2 = (const float*)d_in[12];
  const int*   burn   = (const int*)d_in[13];
  float* out = (float*)d_out;
  char* ws = (char*)d_ws;

  size_t off = 0;
  auto alloc = [&](size_t bytes) -> size_t {
    off = (off + 255) & ~(size_t)255;
    size_t o = off;
    off += bytes;
    return o;
  };

  const size_t o_wih  = alloc((size_t)L_ * G4_ * DH_ * sizeof(bf16_t));
  const size_t o_whh  = alloc((size_t)L_ * G4_ * DH_ * sizeof(bf16_t));
  const size_t o_ew2t = alloc((size_t)H_ * H_ * sizeof(bf16_t));
  const size_t o_dw1t = alloc((size_t)DH_ * DH_ * sizeof(bf16_t));
  const size_t o_dw2t = alloc((size_t)48 * DH_ * sizeof(bf16_t));
  const size_t o_bsum = alloc((size_t)L_ * G4_ * sizeof(float));
  const size_t o_xenc = alloc((size_t)B_ * DH_ * sizeof(bf16_t));
  const size_t o_h    = alloc((size_t)L_ * 2 * B_ * DH_ * sizeof(bf16_t)); // [l][parity]
  const size_t o_c    = alloc((size_t)L_ * B_ * DH_ * sizeof(float));
  const size_t o_y1   = alloc((size_t)B_ * DH_ * sizeof(bf16_t));
  const size_t o_cin  = alloc((size_t)B_ * A_ * DM_ * sizeof(float));
  if (off > ws_size) return;  // workspace too small; bail deterministically

  bf16_t* wihb = (bf16_t*)(ws + o_wih);
  bf16_t* whhb = (bf16_t*)(ws + o_whh);
  bf16_t* ew2t = (bf16_t*)(ws + o_ew2t);
  bf16_t* dw1t = (bf16_t*)(ws + o_dw1t);
  bf16_t* dw2t = (bf16_t*)(ws + o_dw2t);
  float*  bsum = (float*)(ws + o_bsum);
  bf16_t* xenc = (bf16_t*)(ws + o_xenc);
  bf16_t* hbuf = (bf16_t*)(ws + o_h);
  float*  cbuf = (float*)(ws + o_c);
  bf16_t* y1   = (bf16_t*)(ws + o_y1);
  float*  cin  = (float*)(ws + o_cin);

  // ---- one-time packing / init ----
  pack_bf16_kernel<<<2048, 256, 0, stream>>>(w_ih, wihb, (size_t)L_ * G4_ * DH_);
  pack_bf16_kernel<<<2048, 256, 0, stream>>>(w_hh, whhb, (size_t)L_ * G4_ * DH_);
  pack_transpose_kernel<<<64, 256, 0, stream>>>(enc_w2, ew2t, H_, H_, H_);
  pack_transpose_kernel<<<1024, 256, 0, stream>>>(dec_w1, dw1t, DH_, DH_, DH_);
  pack_transpose_kernel<<<256, 256, 0, stream>>>(dec_w2, dw2t, DH_, A_ * DM_, 48);
  bias_sum_kernel<<<(L_ * G4_ + 255) / 256, 256, 0, stream>>>(b_ih, b_hh, bsum, L_ * G4_);
  zero_kernel<<<512, 256, 0, stream>>>((uint32_t*)hbuf,
                                       (size_t)L_ * 2 * B_ * DH_ * sizeof(bf16_t) / 4);
  zero_kernel<<<512, 256, 0, stream>>>((uint32_t*)cbuf,
                                       (size_t)L_ * B_ * DH_ * sizeof(float) / 4);

  const size_t hstride = (size_t)B_ * DH_;  // one [128][1280] h buffer

  // ---- 63 sequential recurrent steps ----
  for (int t = 0; t < NT_; ++t) {
    const int rp = t & 1, wp = rp ^ 1;      // h read/write parity

    // encoder performs burn-in select (device-side) + both encoder MLP layers
    encoder_kernel<<<10, 256, 0, stream>>>(inputs, burn, cin, enc_w1, enc_b1,
                                           ew2t, enc_b2, xenc, t);

    // layer 0
    lstm_layer_kernel<<<20, 256, 0, stream>>>(
        xenc, hbuf + (0 * 2 + rp) * hstride,
        wihb + (size_t)0 * G4_ * DH_, whhb + (size_t)0 * G4_ * DH_,
        bsum + 0 * G4_, cbuf + (size_t)0 * B_ * DH_,
        hbuf + (0 * 2 + wp) * hstride);
    // layer 1 (input = layer 0's new h)
    lstm_layer_kernel<<<20, 256, 0, stream>>>(
        hbuf + (0 * 2 + wp) * hstride, hbuf + (1 * 2 + rp) * hstride,
        wihb + (size_t)1 * G4_ * DH_, whhb + (size_t)1 * G4_ * DH_,
        bsum + 1 * G4_, cbuf + (size_t)1 * B_ * DH_,
        hbuf + (1 * 2 + wp) * hstride);

    dec1_kernel<<<10, 256, 0, stream>>>(hbuf + (1 * 2 + wp) * hstride, dw1t, dec_b1, y1);
    dec2_kernel<<<1, 256, 0, stream>>>(y1, dw2t, dec_b2, cin, out, t);
  }
}
